// KernelDensityEstimator_28363964023355
// MI455X (gfx1250) — matrix operand order repaired and checked by reading, not compiled, not run on hardware
//
#include <hip/hip_runtime.h>
#include <math.h>

typedef _Float16 v16h __attribute__((ext_vector_type(16)));
typedef _Float16 v8h  __attribute__((ext_vector_type(8)));
typedef float    v8f  __attribute__((ext_vector_type(8)));
typedef unsigned int v4u __attribute__((ext_vector_type(4)));
typedef int          v4i __attribute__((ext_vector_type(4)));
typedef int          v8i __attribute__((ext_vector_type(8)));

constexpr int KDIM = 512;        // feature dimension
constexpr int LDSW = KDIM + 8;   // padded LDS row stride (halfs) -> no bank conflicts

union V16 { v16h v; v8h h[2]; };

__device__ __forceinline__ v8f wmma_f16(v16h a, v16h b, v8f c) {
    // D = A(16x32 f16) * B(32x16 f16) + C(16x16 f32)
    return __builtin_amdgcn_wmma_f32_16x16x32_f16(
        /*neg_a=*/false, a, /*neg_b=*/false, b,
        /*c_mod=*/(short)0, c, /*reuse_a=*/false, /*reuse_b=*/false);
}

// ---------------------------------------------------------------------------
// Aux kernel 1: split f32 -> f16 hi + f16 lo  (v = hi + lo, ~22-bit mantissa)
// ---------------------------------------------------------------------------
__global__ void split_f32_kernel(const float* __restrict__ in,
                                 _Float16* __restrict__ hi,
                                 _Float16* __restrict__ lo, int n) {
    for (int i = blockIdx.x * blockDim.x + threadIdx.x; i < n;
         i += gridDim.x * blockDim.x) {
        float v = in[i];
        _Float16 h = (_Float16)v;
        hi[i] = h;
        lo[i] = (_Float16)(v - (float)h);
    }
}

// ---------------------------------------------------------------------------
// Aux kernel 2: row squared norms (one wave32 per row, float4 loads)
// ---------------------------------------------------------------------------
__global__ void rownorm_kernel(const float* __restrict__ a,
                               float* __restrict__ out, int nrows) {
    const int wave = threadIdx.x >> 5;
    const int lane = threadIdx.x & 31;
    const int row  = blockIdx.x * 8 + wave;
    if (row >= nrows) return;
    const float4* p = (const float4*)(a + (size_t)row * KDIM);
    float s = 0.0f;
#pragma unroll
    for (int c = 0; c < KDIM / 4 / 32; ++c) {
        float4 v = p[lane + c * 32];
        s += v.x * v.x + v.y * v.y + v.z * v.z + v.w * v.w;
    }
#pragma unroll
    for (int off = 16; off >= 1; off >>= 1) s += __shfl_xor(s, off, 32);
    if (lane == 0) out[row] = s;
}

// ---------------------------------------------------------------------------
// Main fused KDE kernel: 16 test rows per workgroup, 8 waves, flash-logsumexp
// ---------------------------------------------------------------------------
__global__ __launch_bounds__(256)
void kde_main_kernel(const float* __restrict__ x,
                     const _Float16* __restrict__ th,   // train hi (f16)
                     const _Float16* __restrict__ tl,   // train lo (f16)
                     const float* __restrict__ t2g,     // train row norms
                     const float* __restrict__ x2g,     // test row norms
                     float* __restrict__ out, int n_train) {
    __shared__ float    xs_raw[16 * KDIM];   // 32 KB raw f32 staging (TDM dest)
    __shared__ _Float16 xs_hi[16 * LDSW];
    __shared__ _Float16 xs_lo[16 * LDSW];
    __shared__ float x2s[16];
    __shared__ float red_m[8][16];
    __shared__ float red_s[8][16];

    const int tid = threadIdx.x;
    const int i0  = blockIdx.x * 16;

    // ---- Stage raw x tile (16 x 512 f32) into LDS via the Tensor Data Mover.
#if __has_builtin(__builtin_amdgcn_tensor_load_to_lds) && \
    __has_builtin(__builtin_amdgcn_s_wait_tensorcnt)
    if (tid < 32) {                          // wave 0 issues the TDM op
        const unsigned long long ga =
            (unsigned long long)(uintptr_t)(x + (size_t)i0 * KDIM);
        const unsigned int lds_addr =
            (unsigned int)(size_t)(&xs_raw[0]); // low 32 bits = LDS byte offset
        // D# group 0: count=1 | lds_addr | global_addr | type=2 ("image")
        v4u g0;
        g0[0] = 1u;
        g0[1] = lds_addr;
        g0[2] = (unsigned int)ga;
        g0[3] = (unsigned int)((ga >> 32) & 0x01FFFFFFull) | (2u << 30);
        // D# group 1: data_size=4B; tensor 512x16; tile 512x16; stride 512
        v8i g1;
        g1[0] = (int)(2u << 16);             // [17:16] data_size = 2 (4 bytes)
        g1[1] = (int)(512u << 16);           // [63:48]  tensor_dim0 lo = 512
        g1[2] = (int)(16u << 16);            // [95:80]  tensor_dim1 lo = 16
        g1[3] = (int)(512u << 16);           // [127:112] tile_dim0 = 512
        g1[4] = 16;                          // [143:128] tile_dim1 = 16
        g1[5] = 512;                         // [191:160] tensor_dim0_stride
        g1[6] = 0;
        g1[7] = 0;
        v4i z4 = {0, 0, 0, 0};               // groups 2/3 unused (2D tensor)
#if defined(__clang_major__) && (__clang_major__ >= 23)
        v8i z8 = {0, 0, 0, 0, 0, 0, 0, 0};
        __builtin_amdgcn_tensor_load_to_lds(g0, g1, z4, z4, z8, 0);
#else
        __builtin_amdgcn_tensor_load_to_lds(g0, g1, z4, z4, 0);
#endif
        __builtin_amdgcn_s_wait_tensorcnt(0);
    }
#else
    for (int idx = tid; idx < 16 * KDIM; idx += 256) {
        int row = idx >> 9, c = idx & (KDIM - 1);
        xs_raw[idx] = x[(size_t)(i0 + row) * KDIM + c];
    }
#endif
    __syncthreads();

    // ---- Convert staged f32 -> split f16 hi/lo (padded layout).
    for (int idx = tid; idx < 16 * KDIM; idx += 256) {
        int row = idx >> 9, c = idx & (KDIM - 1);
        float v = xs_raw[idx];
        _Float16 h = (_Float16)v;
        xs_hi[row * LDSW + c] = h;
        xs_lo[row * LDSW + c] = (_Float16)(v - (float)h);
    }
    if (tid < 16) x2s[tid] = x2g[i0 + tid];
    __syncthreads();

    const int wave = tid >> 5;
    const int lane = tid & 31;
    const int col  = lane & 15;       // C-matrix: lane -> N column
    const int hi16 = lane >> 4;       // lane half
    const int ac   = hi16 ? 16 : 0;   // A-fragment K sub-offset per lane half
    const int bc   = hi16 ? 16 : 0;   // B-fragment K sub-offset per lane half

    // Hoist A_hi fragments for all 16 K-chunks (reused across every tile).
    V16 afr[16];
#pragma unroll
    for (int kc = 0; kc < 16; ++kc) {
        const int k0 = kc * 32;
        afr[kc].h[0] = *(const v8h*)&xs_hi[col * LDSW + k0 + ac];
        afr[kc].h[1] = *(const v8h*)&xs_hi[col * LDSW + k0 + 8 + ac];
    }
    float x2v[8];
#pragma unroll
    for (int r = 0; r < 8; ++r) x2v[r] = x2s[hi16 * 8 + r];

    float m[8], s[8];
#pragma unroll
    for (int r = 0; r < 8; ++r) { m[r] = -INFINITY; s[r] = 0.0f; }

    const float kH = -0.5f * 1.44269504088896340736f;  // -0.5 * log2(e)
    const int ntiles = n_train >> 7;                   // 16 cols * 8 waves

    for (int t = 0; t < ntiles; ++t) {
        const int j0 = (t * 8 + wave) * 16;
        const _Float16* bh = th + (size_t)(j0 + col) * KDIM + bc;
        const _Float16* bl = tl + (size_t)(j0 + col) * KDIM + bc;
        v8f acc = {};
#pragma unroll
        for (int kc = 0; kc < 16; ++kc) {
            const int k0 = kc * 32;
            v16h bhv = *(const v16h*)(bh + k0);  // 2x global_load_b128
            v16h blv = *(const v16h*)(bl + k0);
            V16 al;
            al.h[0] = *(const v8h*)&xs_lo[col * LDSW + k0 + ac];
            al.h[1] = *(const v8h*)&xs_lo[col * LDSW + k0 + 8 + ac];
            acc = wmma_f16(afr[kc].v, bhv, acc);   // hi*hi
            acc = wmma_f16(afr[kc].v, blv, acc);   // hi*lo
            acc = wmma_f16(al.v,      bhv, acc);   // lo*hi
        }
        if (t + 1 < ntiles) {
            const _Float16* pf =
                th + (size_t)(((t + 1) * 8 + wave) * 16 + col) * KDIM + bc;
            __builtin_prefetch(pf, 0, 0);          // global_prefetch_b8
        }
        const float t2v = t2g[j0 + col];
        // Online logsumexp update (log2 domain) for the 8 rows in this lane.
#pragma unroll
        for (int r = 0; r < 8; ++r) {
            float sq = fmaxf(x2v[r] + t2v - 2.0f * acc[r], 0.0f);
            float lv = sq * kH;
            float mn = fmaxf(m[r], lv);
            s[r] = s[r] * exp2f(m[r] - mn) + exp2f(lv - mn);
            m[r] = mn;
        }
    }

    // Butterfly merge across the 16 lanes of each half (same row set).
#pragma unroll
    for (int off = 8; off >= 1; off >>= 1) {
#pragma unroll
        for (int r = 0; r < 8; ++r) {
            float om = __shfl_xor(m[r], off, 32);
            float os = __shfl_xor(s[r], off, 32);
            float mn = fmaxf(m[r], om);
            s[r] = s[r] * exp2f(m[r] - mn) + os * exp2f(om - mn);
            m[r] = mn;
        }
    }
    if (col == 0) {
#pragma unroll
        for (int r = 0; r < 8; ++r) {
            red_m[wave][hi16 * 8 + r] = m[r];
            red_s[wave][hi16 * 8 + r] = s[r];
        }
    }
    __syncthreads();

    // Cross-wave merge + final output.
    if (tid < 16) {
        float M = -INFINITY, S = 0.0f;
#pragma unroll
        for (int w = 0; w < 8; ++w) {
            float mm = red_m[w][tid], ss = red_s[w][tid];
            float mn = fmaxf(M, mm);
            S = S * exp2f(M - mn) + ss * exp2f(mm - mn);
            M = mn;
        }
        const float LN2 = 0.69314718055994530942f;
        // Z = 0.5*512*ln(2*pi) + 512*ln(1) + ln(16384)
        const float Z = 256.0f * 1.83787706640934548356f
                      + 9.70406052783923030309f;
        out[i0 + tid] = (M + __log2f(S)) * LN2 - Z;
    }
}

// ---------------------------------------------------------------------------
extern "C" void kernel_launch(void* const* d_in, const int* in_sizes, int n_in,
                              void* d_out, int out_size, void* d_ws,
                              size_t ws_size, hipStream_t stream) {
    const float* x  = (const float*)d_in[0];   // [n_test, 512]
    const float* tr = (const float*)d_in[1];   // [n_train, 512]
    float* out = (float*)d_out;                // [n_test]

    const int n_test  = in_sizes[0] / KDIM;
    const int n_train = in_sizes[1] / KDIM;

    // Workspace layout: train_hi f16 | train_lo f16 | t2 f32 | x2 f32
    char* ws = (char*)d_ws;
    _Float16* th = (_Float16*)ws;
    _Float16* tl = th + (size_t)n_train * KDIM;
    float* t2 = (float*)(tl + (size_t)n_train * KDIM);
    float* x2 = t2 + n_train;

    split_f32_kernel<<<4096, 256, 0, stream>>>(tr, th, tl, n_train * KDIM);
    rownorm_kernel<<<(n_train + 7) / 8, 256, 0, stream>>>(tr, t2, n_train);
    rownorm_kernel<<<(n_test + 7) / 8, 256, 0, stream>>>(x, x2, n_test);
    kde_main_kernel<<<n_test / 16, 256, 0, stream>>>(x, th, tl, t2, x2, out,
                                                     n_train);
}
